// SVQ_61890478735796
// MI455X (gfx1250) — matrix operand order repaired and checked by reading, not compiled, not run on hardware
//
#include <hip/hip_runtime.h>

// MI455X / gfx1250, wave32. Spherical VQ nearest-code via v_wmma_f32_16x16x32_f16
// with double-buffered GLOBAL_LOAD_ASYNC_TO_LDS staging of the codebook tiles.

typedef __attribute__((ext_vector_type(8)))  _Float16 v8h;
typedef __attribute__((ext_vector_type(16))) _Float16 v16h;
typedef __attribute__((ext_vector_type(8)))  float    v8f;

// 128-bit payload type expected by the async-LDS builtins (per hipcc diagnostic)
typedef int v4i __attribute__((vector_size(16)));

typedef __attribute__((address_space(1))) void gvoid;
typedef __attribute__((address_space(3))) void lvoid;
typedef __attribute__((address_space(1))) v4i* gptr_b128;
typedef __attribute__((address_space(3))) v4i* lptr_b128;

#define B_    16
#define N_    256
#define T_    4096
#define NCODE 1024

#define BROW  264   // padded LDS row stride for B tile (f16): 528B = 33*16B

#if __has_builtin(__builtin_amdgcn_global_load_async_to_lds_b128) && \
    __has_builtin(__builtin_amdgcn_s_wait_asynccnt)
#define HAVE_ASYNC_LDS 1
#else
#define HAVE_ASYNC_LDS 0
#endif

// ---------------------------------------------------------------- emb f32->f16
__global__ __launch_bounds__(256) void svq_cvt_emb(const float* __restrict__ emb,
                                                   _Float16* __restrict__ embh) {
  int i = blockIdx.x * 256 + threadIdx.x;
  embh[i] = (_Float16)emb[i];
}

// ------------------------------------------------- lane xor-swizzle (wave32)
template <int XORMASK>
__device__ __forceinline__ int lane_swizzle_xor(int v) {
  // group-of-32 mode: and_mask=0x1f, or_mask=0, xor_mask=XORMASK
  return __builtin_amdgcn_ds_swizzle(v, (XORMASK << 10) | 0x1f);
}

template <int XORMASK>
__device__ __forceinline__ void argmax_step(float& v, int& i) {
  float ov = __int_as_float(lane_swizzle_xor<XORMASK>(__float_as_int(v)));
  int   oi = lane_swizzle_xor<XORMASK>(i);
  // prefer larger dot; on ties prefer the smaller code index (jnp.argmin rule)
  if (ov > v || (ov == v && oi < i)) { v = ov; i = oi; }
}

// --------------------------------------- async prefetch of one 16-code B tile
// 8 KB payload: 512 chunks of 16B; 256 threads issue 2 async b128 loads each.
__device__ __forceinline__ void prefetch_btile(const _Float16* __restrict__ embh,
                                               int nt, _Float16* dst, int tid) {
#if HAVE_ASYNC_LDS
  #pragma unroll
  for (int j = 0; j < 2; ++j) {
    int q   = j * 256 + tid;    // chunk id 0..511
    int row = q >> 5;           // code row 0..15
    int col = q & 31;           // 16B chunk within row (8 f16)
    const _Float16* g = embh + (size_t)(nt * 16 + row) * N_ + col * 8;
    _Float16*       l = dst + row * BROW + col * 8;   // 16B-aligned (BROW*2=528)
    gvoid* gv = (gvoid*)g;      // addrspacecast generic -> global
    lvoid* lv = (lvoid*)l;      // addrspacecast generic -> LDS
    __builtin_amdgcn_global_load_async_to_lds_b128(
        (gptr_b128)gv, (lptr_b128)lv, /*offset=*/0, /*cpol=*/0);
  }
#endif
}

// ------------------------------------------------------------ main argmax GEMM
// grid = 512 blocks (B*T/128), block = 256 threads = 8 waves.
// Block handles 128 consecutive tokens of one batch b; wave w handles 16 tokens.
__global__ __launch_bounds__(256) void svq_argmax(const float* __restrict__ x,
                                                  const _Float16* __restrict__ embh,
                                                  int* __restrict__ idxout) {
  __shared__ _Float16 xtile[128 * 64];      // [token][k_local], 16 KB (setup only)
  __shared__ _Float16 btile[2][16 * BROW];  // double-buffered B tiles, 2 x 8.25 KB

  const int tid  = threadIdx.x;
  const int blk  = blockIdx.x;        // 0..511
  const int b    = blk >> 5;          // batch
  const int t0   = (blk & 31) << 7;   // token base within batch (*128)
  const int wave = tid >> 5;
  const int lane = tid & 31;
  const int half = lane >> 4;         // 0: lanes 0-15, 1: lanes 16-31
  const int lm   = lane & 15;
  const int m0   = wave << 4;         // wave's first token (local)

  // kick off the first B tile while we stage x
  prefetch_btile(embh, 0, &btile[0][0], tid);

  // ---- stage x[b, :, t0..t0+127] through LDS (transpose + f32->f16),
  //      building the 8 register-resident A fragments (16 tokens x 256 dims).
  v16h afrag[8];
  #pragma unroll
  for (int p = 0; p < 4; ++p) {
    const int k0 = p << 6;            // 64 dims per pass
    __syncthreads();
    #pragma unroll
    for (int i = 0; i < 32; ++i) {
      int linear = i * 256 + tid;     // 0..8191
      int kl = linear >> 7;           // 0..63
      int tl = linear & 127;          // 0..127  (coalesced in t)
      float v = x[(size_t)b * ((size_t)N_ * T_) + (size_t)(k0 + kl) * T_ + (t0 + tl)];
      xtile[tl * 64 + kl] = (_Float16)v;
    }
    __syncthreads();
    // A in this code's contiguous k order (B matches): lanes 0-15 -> K {kb..kb+15} with kb=cc*32;
    // lanes 16-31 -> same pattern shifted by 16.
    #pragma unroll
    for (int cc = 0; cc < 2; ++cc) {
      int kb = cc * 32 + 16 * half;
      const v8h* lo = (const v8h*)&xtile[(m0 + lm) * 64 + kb];        // 16B aligned
      const v8h* hi = (const v8h*)&xtile[(m0 + lm) * 64 + kb + 8];    // 16B aligned
      afrag[p * 2 + cc] = __builtin_shufflevector(
          *lo, *hi, 0, 1, 2, 3, 4, 5, 6, 7, 8, 9, 10, 11, 12, 13, 14, 15);
    }
  }

  // ---- sweep all 1024 codes in 64 tiles of 16, K=256 as 8 chained WMMAs.
  float bestv[8];
  int   besti[8];
  #pragma unroll
  for (int r = 0; r < 8; ++r) { bestv[r] = -3.402823466e38f; besti[r] = 0; }

  #pragma unroll 1
  for (int nt = 0; nt < 64; ++nt) {
#if HAVE_ASYNC_LDS
    // double buffer: issue next tile, wait for current (2 loads may remain in
    // flight = the ones just issued; async loads complete in order).
    prefetch_btile(embh, (nt + 1) & 63, &btile[(nt + 1) & 1][0], tid);
    __builtin_amdgcn_s_wait_asynccnt(2);
    __syncthreads();   // all threads' portions of the current tile are in LDS
    // B in the same contiguous k order as A: lane holds 16 consecutive dims of code (nt*16+lm),
    // starting at K = c*32 + 16*half  -> two aligned ds_load_b128 per fragment.
    const _Float16* bp = &btile[nt & 1][0] + lm * BROW + 16 * half;
#else
    const _Float16* bp = embh + (size_t)(nt * 16 + lm) * N_ + 16 * half;
#endif
    v8f acc = {0.f, 0.f, 0.f, 0.f, 0.f, 0.f, 0.f, 0.f};
    #pragma unroll
    for (int c = 0; c < 8; ++c) {
      v16h bfrag = *(const v16h*)(bp + c * 32);
      acc = __builtin_amdgcn_wmma_f32_16x16x32_f16(false, afrag[c],
                                                   false, bfrag,
                                                   (short)0, acc,
                                                   false, false);
    }
    // C layout: lane holds N = nt*16+lm for rows M = r + 8*half.
    const int code = nt * 16 + lm;
    #pragma unroll
    for (int r = 0; r < 8; ++r) {
      float v = acc[r];
      if (v > bestv[r]) { bestv[r] = v; besti[r] = code; }  // strict > keeps lowest nt
    }
#if HAVE_ASYNC_LDS
    __syncthreads();   // everyone done reading before this buffer is overwritten
#endif
  }

  // ---- reduce across the 16 lanes of each half (they cover code residues 0..15)
  #pragma unroll
  for (int r = 0; r < 8; ++r) {
    argmax_step<8>(bestv[r], besti[r]);
    argmax_step<4>(bestv[r], besti[r]);
    argmax_step<2>(bestv[r], besti[r]);
    argmax_step<1>(bestv[r], besti[r]);
  }

  if (lm == 0) {  // lanes 0 and 16: tokens m0+0..7 and m0+8..15
    int tokbase = b * T_ + t0 + m0 + half * 8;
    #pragma unroll
    for (int r = 0; r < 8; ++r) idxout[tokbase + r] = besti[r];
  }
}

// ----------------------------------------------------- gather + transpose out
// out[b, n, t] = emb_f32[idx[b*T + t], n]; consecutive threads -> consecutive t.
__global__ __launch_bounds__(256) void svq_gather(const float* __restrict__ emb,
                                                  const int* __restrict__ idx,
                                                  float* __restrict__ out) {
  size_t o = (size_t)blockIdx.x * 256 + threadIdx.x;
  int t = (int)(o & (T_ - 1));
  int n = (int)((o >> 12) & (N_ - 1));   // T_ = 2^12
  int b = (int)(o >> 20);                // N_*T_ = 2^20
  out[o] = emb[(size_t)idx[b * T_ + t] * N_ + n];
}

// ---------------------------------------------------------------------- launch
extern "C" void kernel_launch(void* const* d_in, const int* in_sizes, int n_in,
                              void* d_out, int out_size, void* d_ws, size_t ws_size,
                              hipStream_t stream) {
  const float* x   = (const float*)d_in[0];   // [16, 256, 4096] f32
  const float* emb = (const float*)d_in[1];   // [1024, 256] f32
  float* out = (float*)d_out;                 // [16, 256, 4096] f32

  _Float16* embh = (_Float16*)d_ws;                                       // 512 KB
  int* idx = (int*)((char*)d_ws + (size_t)NCODE * N_ * sizeof(_Float16)); // 256 KB

  svq_cvt_emb<<<(NCODE * N_) / 256, 256, 0, stream>>>(emb, embh);
  svq_argmax<<<(B_ * T_) / 128, 256, 0, stream>>>(x, embh, idx);
  svq_gather<<<(int)(((size_t)B_ * N_ * T_) / 256), 256, 0, stream>>>(emb, idx, out);
}
